// SpikingConnectomeLinear_42880953483369
// MI455X (gfx1250) — compile-verified, matrix operand order checked by reading, not run on hardware
//
#include <hip/hip_runtime.h>

#define BATCH  1024
#define N_PRE  8192
#define N_POST 8192

typedef __attribute__((ext_vector_type(16))) _Float16 v16h;
typedef __attribute__((ext_vector_type(8)))  float    v8f;

__device__ __forceinline__ unsigned short f2h_bits(float f) {
  union { _Float16 h; unsigned short u; } cv;
  cv.h = (_Float16)f;
  return cv.u;
}

// CDNA5 async global->LDS copy, 16 bytes per lane (ASYNCcnt-tracked).
__device__ __forceinline__ void async_cp16(unsigned lds_addr, const void* gptr) {
  asm volatile("global_load_async_to_lds_b128 %0, %1, off"
               :: "v"(lds_addr), "v"((unsigned long long)(uintptr_t)gptr)
               : "memory");
}

__device__ __forceinline__ void wait_async0() {
  asm volatile("s_wait_asynccnt 0x0" ::: "memory");
}

// ---------------------------------------------------------------------------
// Phase 1: x1 / U fused elementwise; writes A = spikes*U*x1 as f16 (workspace)
// and x_new directly into the second output slot.
// ---------------------------------------------------------------------------
__global__ __launch_bounds__(256)
void prep_kernel(const float* __restrict__ spikes,
                 const float* __restrict__ x_std,
                 const float* __restrict__ log_tau_rec,
                 const float* __restrict__ logit_U,
                 _Float16* __restrict__ A_h,
                 float* __restrict__ x_new) {
  const float DT = 0.001f;
  const float tau_rec   = __expf(log_tau_rec[0]);
  const float alpha_rec = __expf(-DT / tau_rec);
  const float U = 1.0f / (1.0f + __expf(-logit_U[0]));

  size_t i = ((size_t)blockIdx.x * blockDim.x + threadIdx.x) * 4;
  float4 s4 = *(const float4*)(spikes + i);
  float4 x4 = *(const float4*)(x_std + i);
  float sv[4] = {s4.x, s4.y, s4.z, s4.w};
  float xv[4] = {x4.x, x4.y, x4.z, x4.w};
  unsigned short ah[4];
  float xo[4];
#pragma unroll
  for (int k = 0; k < 4; ++k) {
    float x1 = 1.0f - (1.0f - xv[k]) * alpha_rec;
    ah[k] = f2h_bits(sv[k] * U * x1);                       // A operand (f16)
    float sc = fminf(fmaxf(sv[k], 0.0f), 1.0f);
    xo[k] = x1 * (1.0f - U * sc);                           // x_new output
  }
  uint2 p;
  p.x = (unsigned int)ah[0] | ((unsigned int)ah[1] << 16);
  p.y = (unsigned int)ah[2] | ((unsigned int)ah[3] << 16);
  *(uint2*)(A_h + i) = p;
  *(float4*)(x_new + i) = make_float4(xo[0], xo[1], xo[2], xo[3]);
}

// ---------------------------------------------------------------------------
// Phase 2: Wnm f32 [k][n]  ->  Wt f16 [n][k]  (tiled 32x32 transpose+convert).
// The f16 copy (128 MB) is L2-resident for the GEMM's repeated reads, and the
// [n][k] layout makes the GEMM's B-tile fill a contiguous copy along K
// (async-copyable, no in-register transpose).
// Strength (5e-9) underflows f16, so it is applied in the f32 epilogue.
// ---------------------------------------------------------------------------
__global__ __launch_bounds__(256)
void convw_t_kernel(const float* __restrict__ W, _Float16* __restrict__ Wt) {
  __shared__ float tile[32][33];
  const int tx = threadIdx.x & 31;
  const int ty = threadIdx.x >> 5;           // 0..7
  const int bn = blockIdx.x * 32;            // n tile
  const int bk = blockIdx.y * 32;            // k tile
#pragma unroll
  for (int r = 0; r < 4; ++r) {              // read: coalesced along n
    int k = bk + ty + r * 8;
    tile[ty + r * 8][tx] = W[(size_t)k * N_POST + bn + tx];
  }
  __syncthreads();
#pragma unroll
  for (int r = 0; r < 4; ++r) {              // write: coalesced along k
    int n = bn + ty + r * 8;
    Wt[(size_t)n * N_PRE + bk + tx] = (_Float16)tile[tx][ty + r * 8];
  }
}

// ---------------------------------------------------------------------------
// Phase 3: WMMA GEMM (M=1024, N=8192, K=8192) with fused epilogue:
//   I_out = I_syn*ALPHA_SYN + strength*acc*(1 + noise*0.15)
// Block = 256 threads = 8 wave32 waves in 4(M) x 2(N); each wave owns a 32x64
// sub-tile = 2x4 fragments. K staged 64-wide, DOUBLE-BUFFERED in LDS via
// global_load_async_to_lds_b128; one barrier per chunk; next chunk's copies
// overlap this chunk's 16 WMMAs.
//
// LDS rows are 36 uints (144 B): keeps async B128 stores 16 B aligned and
// r*36 mod 64 hits 16 distinct banks for the 16-lane fragment gathers.
// ---------------------------------------------------------------------------
#define ROWU 36                               // uints per LDS tile row

__global__ __launch_bounds__(256)
void gemm_kernel(const _Float16* __restrict__ Ah,
                 const _Float16* __restrict__ Wt,
                 const float* __restrict__ I_syn,
                 const float* __restrict__ noise,
                 const float* __restrict__ log_strength,
                 float* __restrict__ I_out) {
  __shared__ unsigned int As[2][128][ROWU];   // [m][k-pair], double buffered
  __shared__ unsigned int Bs[2][128][ROWU];   // [n][k-pair], double buffered

  const int tid  = threadIdx.x;
  const int lane = tid & 31;
  const int wave = tid >> 5;
  const int m_sub = (wave >> 1) * 32;
  const int n_sub = (wave & 1) * 64;
  const int nb = blockIdx.x * 128;
  const int mb = blockIdx.y * 128;
  const int l16  = lane & 15;
  const int half = lane >> 4;

  const unsigned asb0 = (unsigned)(uintptr_t)&As[0][0][0];
  const unsigned bsb0 = (unsigned)(uintptr_t)&Bs[0][0][0];
  const unsigned bufBytes = 128u * ROWU * 4u;

  v8f acc[2][4];
#pragma unroll
  for (int i = 0; i < 2; ++i)
#pragma unroll
    for (int j = 0; j < 4; ++j) acc[i][j] = (v8f)(0.0f);

  // per-thread fill coordinates (128 rows x 8 x 16B chunks = 1024 lane-ops)
  const int fr0 = tid >> 3;                   // rows tid/8, +32, +64, +96
  const int fc  = (tid & 7) * 8;              // halves offset within chunk row

  // ---- issue chunk 0 into buffer 0 ----
#pragma unroll
  for (int it = 0; it < 4; ++it) {
    int r = fr0 + it * 32;
    async_cp16(asb0 + (unsigned)(r * (ROWU * 4) + fc * 2),
               Ah + (size_t)(mb + r) * N_PRE + fc);
    async_cp16(bsb0 + (unsigned)(r * (ROWU * 4) + fc * 2),
               Wt + (size_t)(nb + r) * N_PRE + fc);
  }

  int buf = 0;
  for (int kt = 0; kt < N_PRE; kt += 64) {
    wait_async0();                            // our copies for 'buf' done
    __syncthreads();                          // everyone's copies visible

    // ---- overlap: issue next chunk into the other buffer ----
    if (kt + 64 < N_PRE) {
      unsigned asb = asb0 + (unsigned)(buf ^ 1) * bufBytes;
      unsigned bsb = bsb0 + (unsigned)(buf ^ 1) * bufBytes;
      int kn = kt + 64;
#pragma unroll
      for (int it = 0; it < 4; ++it) {
        int r = fr0 + it * 32;
        async_cp16(asb + (unsigned)(r * (ROWU * 4) + fc * 2),
                   Ah + (size_t)(mb + r) * N_PRE + kn + fc);
        async_cp16(bsb + (unsigned)(r * (ROWU * 4) + fc * 2),
                   Wt + (size_t)(nb + r) * N_PRE + kn + fc);
      }
    }

    // ---- two K=32 WMMA steps over the 64-wide chunk ----
#pragma unroll
    for (int ks = 0; ks < 2; ++ks) {
      const int kb2 = ks * 16;                // offset in k-pairs
      v16h af[2], bf[4];
      // A frags: lane holds row M=l16 (+8K for lanes 16-31);
      // VGPR v -> K pair (v<4 ? v : v+4) + 4*half
#pragma unroll
      for (int i = 0; i < 2; ++i) {
        union { v16h h; unsigned int u[8]; } t;
        int row = m_sub + 16 * i + l16;
#pragma unroll
        for (int v = 0; v < 8; ++v) {
          int p = ((v < 4) ? v : v + 4) + (half << 2);
          t.u[v] = As[buf][row][kb2 + p];
        }
        af[i] = t.h;
      }
#pragma unroll
      for (int j = 0; j < 4; ++j) {
        union { v16h h; unsigned int u[8]; } t;
        int col = n_sub + 16 * j + l16;
#pragma unroll
        for (int v = 0; v < 8; ++v) {
          int p = ((v < 4) ? v : v + 4) + (half << 2);
          t.u[v] = Bs[buf][col][kb2 + p];
        }
        bf[j] = t.h;
      }
#pragma unroll
      for (int i = 0; i < 2; ++i)
#pragma unroll
        for (int j = 0; j < 4; ++j)
          acc[i][j] = __builtin_amdgcn_wmma_f32_16x16x32_f16(
              false, af[i], false, bf[j], (short)0, acc[i][j], false, false);
    }
    buf ^= 1;
  }

  // ---- fused epilogue: strength scale + multiplicative noise + I_syn decay ----
  const float ALPHA_SYN = 0.81873075307798182f;   // exp(-0.001/0.005)
  const float strength  = __expf(log_strength[0]);
  const int mhalf = half * 8;                      // C/D: VGPR r -> M=r / M=8+r
#pragma unroll
  for (int i = 0; i < 2; ++i) {
#pragma unroll
    for (int j = 0; j < 4; ++j) {
      int n = nb + n_sub + 16 * j + l16;
#pragma unroll
      for (int r = 0; r < 8; ++r) {
        int m = mb + m_sub + 16 * i + r + mhalf;
        size_t idx = (size_t)m * N_POST + n;
        float inj = strength * acc[i][j][r];
        I_out[idx] = I_syn[idx] * ALPHA_SYN + inj * (1.0f + noise[idx] * 0.15f);
      }
    }
  }
}

// ---------------------------------------------------------------------------
extern "C" void kernel_launch(void* const* d_in, const int* in_sizes, int n_in,
                              void* d_out, int out_size, void* d_ws, size_t ws_size,
                              hipStream_t stream) {
  const float* spikes       = (const float*)d_in[0];
  const float* I_syn        = (const float*)d_in[1];
  const float* x_std        = (const float*)d_in[2];
  const float* noise        = (const float*)d_in[3];
  const float* Wnm          = (const float*)d_in[4];
  const float* log_strength = (const float*)d_in[5];
  const float* log_tau_rec  = (const float*)d_in[6];
  const float* logit_U      = (const float*)d_in[7];

  float* I_out = (float*)d_out;                              // output 0
  float* x_new = I_out + (size_t)BATCH * N_POST;             // output 1

  // workspace: A_f16 (16 MB) then Wt_f16 (128 MB, [n][k], L2-resident)
  _Float16* A_h = (_Float16*)d_ws;
  _Float16* Wt  = (_Float16*)((char*)d_ws + (size_t)BATCH * N_PRE * sizeof(_Float16));

  {
    size_t n = (size_t)BATCH * N_PRE;                        // 8M elems, 4/thread
    prep_kernel<<<(unsigned)(n / 4 / 256), 256, 0, stream>>>(
        spikes, x_std, log_tau_rec, logit_U, A_h, x_new);
  }
  {
    dim3 grid(N_POST / 32, N_PRE / 32);                      // 32x32 transpose tiles
    convw_t_kernel<<<grid, 256, 0, stream>>>(Wnm, Wt);
  }
  {
    dim3 grid(N_POST / 128, BATCH / 128);                    // 64 x 8 tiles
    gemm_kernel<<<grid, 256, 0, stream>>>(A_h, Wt, I_syn, noise, log_strength, I_out);
  }
}
